// Attention_2645699854953
// MI455X (gfx1250) — compile-verified, matrix operand order checked by reading, not run on hardware
//
#include <hip/hip_runtime.h>

// Problem constants (match reference)
#define DIM_    4096
#define NH_     32
#define NKV_    8
#define HD_     128
#define B_      2
#define T_      2048
#define MROWS_  (B_*T_)   // 4096 = B*T

typedef __attribute__((ext_vector_type(16))) __bf16 v16bf;
typedef __attribute__((ext_vector_type(8)))  float  v8f;
typedef __attribute__((ext_vector_type(4)))  unsigned int v4u;
typedef __attribute__((ext_vector_type(8)))  int    v8i;
typedef __attribute__((ext_vector_type(4)))  int    v4i;

__device__ __forceinline__ unsigned short f2bf(float f) {
    unsigned int u = __builtin_bit_cast(unsigned int, f);
    u += 0x7fffu + ((u >> 16) & 1u);   // round-to-nearest-even
    return (unsigned short)(u >> 16);
}
__device__ __forceinline__ float bf2f(unsigned short b) {
    unsigned int u = ((unsigned int)b) << 16;
    return __builtin_bit_cast(float, u);
}

union Frag16 { v16bf v; uint4 q[2]; unsigned short s[16]; };

__device__ __forceinline__ v8f wmma_bf16(const Frag16& a, const Frag16& b, v8f c) {
    return __builtin_amdgcn_wmma_f32_16x16x32_bf16(false, a.v, false, b.v,
                                                   (short)0, c, false, false);
}

// ---------------------------------------------------------------------------
// Tensor Data Mover: 2D bf16 tile (tile_w x tile_h elems, row stride `stride`
// elems in global) -> LDS at lds_addr, with optional LDS row padding so the
// staged layout matches our padded fragment layout exactly.
// pad_interval_code: pad after 2<<code DWORDs; pad_amount_code: insert
// (code+1) DWORDs.  Issue from ONE wave only (TDM ignores EXEC); follow with
// s_wait_tensorcnt before the workgroup barrier.
// ---------------------------------------------------------------------------
__device__ __forceinline__ void tdm_load_2d_bf16(unsigned lds_addr, const void* gaddr,
                                                 unsigned tile_w, unsigned tile_h,
                                                 unsigned stride,
                                                 unsigned pad_interval_code,
                                                 unsigned pad_amount_code) {
    const unsigned long long ga = (unsigned long long)gaddr;
    v4u g0;
    g0[0] = 1u;                                            // count=1 (valid)
    g0[1] = lds_addr;                                      // lds_addr[31:0]
    g0[2] = (unsigned)ga;                                  // global_addr[31:0]
    g0[3] = (unsigned)((ga >> 32) & 0x01FFFFFFull) | (2u << 30); // addr[56:32], type=2
    v8i g1;
    unsigned w0 = (1u << 16);                              // data_size=1 -> 2 bytes
    w0 |= (1u << 20) | (pad_interval_code << 22) | (pad_amount_code << 25); // pad_enable
    g1[0] = (int)w0;
    g1[1] = (int)(tile_w << 16);   // tensor_dim0[15:0] = tile_w (atomic_barrier_addr=0)
    g1[2] = (int)(tile_h << 16);   // tensor_dim0 hi = 0 ; tensor_dim1[15:0] = tile_h
    g1[3] = (int)(tile_w << 16);   // tensor_dim1 hi = 0 ; tile_dim0 = tile_w
    g1[4] = (int)tile_h;           // tile_dim1 = tile_h ; tile_dim2 = 0
    g1[5] = (int)stride;           // tensor_dim0_stride[31:0] (elements)
    g1[6] = 0;                     // stride hi, dim1_stride lo
    g1[7] = 0;
    const v4i gz = {0, 0, 0, 0};
#if defined(__clang_major__) && (__clang_major__ >= 23)
    const v8i gz8 = {0, 0, 0, 0, 0, 0, 0, 0};
    __builtin_amdgcn_tensor_load_to_lds(g0, g1, gz, gz, gz8, 0);
#else
    __builtin_amdgcn_tensor_load_to_lds(g0, g1, gz, gz, 0);
#endif
}

// ---------------------------------------------------------------------------
// Tiled GEMM: C[M,N] = A[M,K] * B[K,N].  A is f32 (staged with f32->bf16
// convert) or bf16 (staged by the Tensor Data Mover); B is f32 converted to
// bf16 while staging transposed; C is f32 or bf16.
// Block = 256 threads (8 waves), 128x128 C tile, K-step 32.
// LDS: A as [m][k] stride 40, B transposed as [n][k] stride 40 (16B aligned,
// bank-conflict-free b128 fragment reads).
// ---------------------------------------------------------------------------
template<bool A_IS_F32, bool OUT_F32>
__global__ __launch_bounds__(256) void gemm_wmma(const void* __restrict__ Ap,
                                                 const float* __restrict__ Bw,
                                                 void* __restrict__ Cp,
                                                 int M, int N, int K) {
    __shared__ unsigned short lA[128 * 40];
    __shared__ unsigned short lB[128 * 40];

    const int tid  = threadIdx.x;
    const int row0 = blockIdx.y * 128;
    const int col0 = blockIdx.x * 128;

    const int wid  = tid >> 5;
    const int lane = tid & 31;
    const int lm   = lane & 15;
    const int hi   = lane >> 4;
    const int mB   = (wid & 3) * 32;   // wave M slice
    const int nB   = (wid >> 2) * 64;  // wave N slice

    const v8f vzero = {0.f,0.f,0.f,0.f,0.f,0.f,0.f,0.f};
    v8f acc[2][4];
    #pragma unroll
    for (int i = 0; i < 2; ++i)
        #pragma unroll
        for (int j = 0; j < 4; ++j) acc[i][j] = vzero;

    const int ar = tid >> 1;           // A stage: row 0..127
    const int ac = (tid & 1) * 16;     // A stage: col block 0/16
    const int bk = tid & 31;           // B stage: k 0..31
    const int bn = (tid >> 5) * 16;    // B stage: n block

    for (int kk = 0; kk < K; kk += 32) {
        // ---- stage A tile 128x32 -> lA[m][k] (stride 40)
        if constexpr (A_IS_F32) {
            Frag16 u;
            const float* src = (const float*)Ap + (size_t)(row0 + ar) * K + kk + ac;
            #pragma unroll
            for (int j = 0; j < 4; ++j) {
                float4 f = ((const float4*)src)[j];
                u.s[j*4+0] = f2bf(f.x); u.s[j*4+1] = f2bf(f.y);
                u.s[j*4+2] = f2bf(f.z); u.s[j*4+3] = f2bf(f.w);
            }
            *(uint4*)&lA[ar * 40 + ac]     = u.q[0];
            *(uint4*)&lA[ar * 40 + ac + 8] = u.q[1];
        } else {
            // bf16 A: Tensor Data Mover copies 128 rows x 32 elems; LDS pad of
            // 4 DWORDs after every 16 DWORDs (one row) -> stride 40 ushorts.
            if (wid == 0)
                tdm_load_2d_bf16((unsigned)(unsigned long long)(void*)&lA[0],
                                 (const unsigned short*)Ap + (size_t)row0 * K + kk,
                                 /*tile_w=*/32, /*tile_h=*/128, /*stride=*/(unsigned)K,
                                 /*pad_interval(16 dw)=*/3, /*pad_amount(4 dw)=*/3);
        }
        // ---- stage B tile 32x128, transposed -> lB[n][k] (stride 40)
        {
            const float* src = Bw + (size_t)(kk + bk) * N + col0 + bn;
            #pragma unroll
            for (int j = 0; j < 4; ++j) {
                float4 f = ((const float4*)src)[j];
                lB[(bn + j*4 + 0) * 40 + bk] = f2bf(f.x);
                lB[(bn + j*4 + 1) * 40 + bk] = f2bf(f.y);
                lB[(bn + j*4 + 2) * 40 + bk] = f2bf(f.z);
                lB[(bn + j*4 + 3) * 40 + bk] = f2bf(f.w);
            }
            if (kk + 32 < K)  // gfx1250 global_prefetch_b8
                __builtin_prefetch(Bw + (size_t)(kk + 32 + bk) * N + col0 + bn, 0, 1);
        }
        if constexpr (!A_IS_F32) {
            if (wid == 0) __builtin_amdgcn_s_wait_tensorcnt(0);
        }
        __syncthreads();

        Frag16 af[2];
        #pragma unroll
        for (int tm = 0; tm < 2; ++tm) {
            const int r = mB + tm * 16 + lm;
            af[tm].q[0] = *(const uint4*)&lA[r * 40 + hi * 8];
            af[tm].q[1] = *(const uint4*)&lA[r * 40 + 16 + hi * 8];
        }
        #pragma unroll
        for (int tn = 0; tn < 4; ++tn) {
            Frag16 bf;
            const int n = nB + tn * 16 + lm;
            bf.q[0] = *(const uint4*)&lB[n * 40 + hi * 8];
            bf.q[1] = *(const uint4*)&lB[n * 40 + 16 + hi * 8];
            #pragma unroll
            for (int tm = 0; tm < 2; ++tm)
                acc[tm][tn] = wmma_bf16(af[tm], bf, acc[tm][tn]);
        }
        __syncthreads();
    }

    // ---- epilogue: C/D layout: VGPR r -> (M=r + hi*8, N=lane&15)
    #pragma unroll
    for (int tm = 0; tm < 2; ++tm)
        #pragma unroll
        for (int tn = 0; tn < 4; ++tn)
            #pragma unroll
            for (int r = 0; r < 8; ++r) {
                const int gm = row0 + mB + tm * 16 + r + hi * 8;
                const int gn = col0 + nB + tn * 16 + lm;
                const float v = acc[tm][tn][r];
                if constexpr (OUT_F32)
                    ((float*)Cp)[(size_t)gm * N + gn] = v;
                else
                    ((unsigned short*)Cp)[(size_t)gm * N + gn] = f2bf(v);
            }
}

// ---------------------------------------------------------------------------
// RoPE in place on bf16 q ([M,4096]) and k ([M,1024]); one thread per pair.
// ---------------------------------------------------------------------------
__global__ __launch_bounds__(256) void rope_kernel(unsigned short* __restrict__ qb,
                                                   unsigned short* __restrict__ kb,
                                                   const float* __restrict__ freqs) {
    const long long idx = (long long)blockIdx.x * 256 + threadIdx.x;
    const long long PQ  = (long long)MROWS_ * NH_ * (HD_ / 2);
    unsigned short* buf;
    long long row; int rem, strideN;
    if (idx < PQ) {
        buf = qb; strideN = NH_ * HD_;
        row = idx / (NH_ * (HD_/2));
        rem = (int)(idx % (NH_ * (HD_/2)));
    } else {
        const long long j = idx - PQ;
        buf = kb; strideN = NKV_ * HD_;
        row = j / (NKV_ * (HD_/2));
        rem = (int)(j % (NKV_ * (HD_/2)));
    }
    const int head = rem >> 6;        // /64
    const int i    = rem & 63;
    const int t    = (int)(row & (T_ - 1));
    const float c  = freqs[((size_t)t * 64 + i) * 2 + 0];
    const float s  = freqs[((size_t)t * 64 + i) * 2 + 1];
    const size_t off = (size_t)row * strideN + head * HD_ + 2 * i;
    const float e = bf2f(buf[off]);
    const float o = bf2f(buf[off + 1]);
    buf[off]     = f2bf(e * c - o * s);
    buf[off + 1] = f2bf(o * c + e * s);
}

// ---------------------------------------------------------------------------
// Flash attention (causal, GQA rep=4).  Block = 128 threads (4 waves) handles
// one (b, h, 64-query tile); each wave owns 16 query rows.  K chunk staged by
// the Tensor Data Mover (pad -> stride 136); V chunk staged transposed
// manually.  All matmuls via v_wmma bf16.
// ---------------------------------------------------------------------------
__global__ __launch_bounds__(128) void attn_wmma(const unsigned short* __restrict__ qb,
                                                 const unsigned short* __restrict__ kb,
                                                 const unsigned short* __restrict__ vb,
                                                 unsigned short* __restrict__ yb) {
    __shared__ unsigned short lK[32 * 136];    // [key][d], TDM-padded rows
    __shared__ unsigned short lVT[128 * 40];   // [d][key]
    __shared__ unsigned short lP[4][16 * 40];  // per-wave P bounce [m][k]

    const int qt = blockIdx.x & 31;
    const int h  = (blockIdx.x >> 5) & 31;
    const int b  = blockIdx.x >> 10;
    const int q0 = qt * 64;
    const int kvh = h >> 2;                    // repeat(k,4): kv head = h/4

    const int tid  = threadIdx.x;
    const int wid  = tid >> 5;
    const int lane = tid & 31;
    const int lm   = lane & 15;
    const int hi   = lane >> 4;
    const int qrow_base = q0 + wid * 16;

    // Q fragments for this wave's 16 rows (16 x 128 bf16 -> 4 A-frags)
    Frag16 qf[4];
    {
        const unsigned short* qrow =
            qb + (size_t)(b * T_ + qrow_base + lm) * (NH_ * HD_) + h * HD_;
        #pragma unroll
        for (int c = 0; c < 4; ++c) {
            qf[c].q[0] = *(const uint4*)(qrow + c * 32 + hi * 8);
            qf[c].q[1] = *(const uint4*)(qrow + c * 32 + 16 + hi * 8);
        }
    }

    const v8f vzero = {0.f,0.f,0.f,0.f,0.f,0.f,0.f,0.f};
    v8f   Oa[8];
    float mS[8], lS[8];
    #pragma unroll
    for (int i = 0; i < 8; ++i) { Oa[i] = vzero; mS[i] = -1e30f; lS[i] = 0.f; }

    const float scale = 0.08838834764831845f;  // 1/sqrt(128)

    for (int kt = 0; kt < q0 + 64; kt += 32) {
        // ---- stage K chunk [32][128] via TDM (wave 0 issues; EXEC ignored by
        // TDM so the branch must be wave-uniform).  LDS pad 4 DWORDs after
        // every 64 DWORDs (one 128-elem row) -> stride 136 ushorts.
        if (wid == 0)
            tdm_load_2d_bf16((unsigned)(unsigned long long)(void*)&lK[0],
                             kb + (size_t)(b * T_ + kt) * (NKV_ * HD_) + kvh * HD_,
                             /*tile_w=*/128, /*tile_h=*/32, /*stride=*/NKV_ * HD_,
                             /*pad_interval(64 dw)=*/5, /*pad_amount(4 dw)=*/3);
        // ---- stage V chunk transposed -> lVT[d][key]
        {
            const int key = tid & 31, db = (tid >> 5) * 32;
            const unsigned short* src =
                vb + (size_t)(b * T_ + kt + key) * (NKV_ * HD_) + kvh * HD_ + db;
            union { uint4 q[4]; unsigned short s[32]; } u;
            #pragma unroll
            for (int j = 0; j < 4; ++j) u.q[j] = ((const uint4*)src)[j];
            #pragma unroll
            for (int j = 0; j < 32; ++j) lVT[(db + j) * 40 + key] = u.s[j];
        }
        if (wid == 0) __builtin_amdgcn_s_wait_tensorcnt(0);
        __syncthreads();

        // ---- S = Q * K^T  (16 x 32 of scores, 2 n-subtiles)
        v8f S[2] = { vzero, vzero };
        #pragma unroll
        for (int nt = 0; nt < 2; ++nt) {
            const int n = nt * 16 + lm;
            #pragma unroll
            for (int c = 0; c < 4; ++c) {
                Frag16 kf;
                kf.q[0] = *(const uint4*)&lK[n * 136 + c * 32 + hi * 8];
                kf.q[1] = *(const uint4*)&lK[n * 136 + c * 32 + 16 + hi * 8];
                S[nt] = wmma_bf16(qf[c], kf, S[nt]);
            }
        }
        // ---- scale + causal mask
        #pragma unroll
        for (int nt = 0; nt < 2; ++nt)
            #pragma unroll
            for (int r = 0; r < 8; ++r) {
                float v = S[nt][r] * scale;
                const int key_idx = kt + nt * 16 + lm;
                const int q_idx   = qrow_base + r + hi * 8;
                S[nt][r] = (key_idx > q_idx) ? -1e30f : v;
            }
        // ---- online softmax (row reductions across 16-lane halves)
        #pragma unroll
        for (int r = 0; r < 8; ++r) {
            float mx = fmaxf(S[0][r], S[1][r]);
            #pragma unroll
            for (int off = 8; off >= 1; off >>= 1)
                mx = fmaxf(mx, __shfl_xor(mx, off));
            const float nm = fmaxf(mS[r], mx);
            const float corr = __expf(mS[r] - nm);
            const float e0 = __expf(S[0][r] - nm);
            const float e1 = __expf(S[1][r] - nm);
            S[0][r] = e0; S[1][r] = e1;
            float rs = e0 + e1;
            #pragma unroll
            for (int off = 8; off >= 1; off >>= 1)
                rs += __shfl_xor(rs, off);
            lS[r] = lS[r] * corr + rs;
            mS[r] = nm;
            #pragma unroll
            for (int dd = 0; dd < 8; ++dd) Oa[dd][r] *= corr;
        }
        // ---- P (C layout) -> bf16 A-frag via per-wave LDS bounce
        #pragma unroll
        for (int nt = 0; nt < 2; ++nt)
            #pragma unroll
            for (int r = 0; r < 8; ++r)
                lP[wid][(r + hi * 8) * 40 + nt * 16 + lm] = f2bf(S[nt][r]);
        asm volatile("s_wait_dscnt 0" ::: "memory");  // wave-local LDS RAW
        Frag16 pf;
        pf.q[0] = *(const uint4*)&lP[wid][lm * 40 + hi * 8];
        pf.q[1] = *(const uint4*)&lP[wid][lm * 40 + 16 + hi * 8];
        // ---- O += P * V
        #pragma unroll
        for (int dd = 0; dd < 8; ++dd) {
            Frag16 vf;
            const int n = dd * 16 + lm;
            vf.q[0] = *(const uint4*)&lVT[n * 40 + hi * 8];
            vf.q[1] = *(const uint4*)&lVT[n * 40 + 16 + hi * 8];
            Oa[dd] = wmma_bf16(pf, vf, Oa[dd]);
        }
        __syncthreads();
    }

    // ---- normalize and write y as bf16 [b*T+t][h*128+d]
    #pragma unroll
    for (int r = 0; r < 8; ++r) {
        const float inv = 1.0f / lS[r];
        const size_t row = (size_t)(b * T_ + qrow_base + r + hi * 8);
        #pragma unroll
        for (int dd = 0; dd < 8; ++dd)
            yb[row * DIM_ + h * HD_ + dd * 16 + lm] = f2bf(Oa[dd][r] * inv);
    }
}

// ---------------------------------------------------------------------------
extern "C" void kernel_launch(void* const* d_in, const int* in_sizes, int n_in,
                              void* d_out, int out_size, void* d_ws, size_t ws_size,
                              hipStream_t stream) {
    (void)in_sizes; (void)n_in; (void)out_size; (void)ws_size;
    const float* x     = (const float*)d_in[0];
    const float* freqs = (const float*)d_in[1];
    const float* wq    = (const float*)d_in[2];
    const float* wk    = (const float*)d_in[3];
    const float* wv    = (const float*)d_in[4];
    const float* wo    = (const float*)d_in[5];
    float* out = (float*)d_out;

    // workspace (bf16): q 32MB + k 8MB + v 8MB + y 32MB ~= 84MB
    unsigned short* qb = (unsigned short*)d_ws;
    unsigned short* kb = qb + (size_t)MROWS_ * (NH_ * HD_);
    unsigned short* vb = kb + (size_t)MROWS_ * (NKV_ * HD_);
    unsigned short* yb = vb + (size_t)MROWS_ * (NKV_ * HD_);

    // QKV projections (f32 in -> bf16 out)
    gemm_wmma<true, false><<<dim3((NH_ * HD_) / 128, MROWS_ / 128), 256, 0, stream>>>(
        x, wq, qb, MROWS_, NH_ * HD_, DIM_);
    gemm_wmma<true, false><<<dim3((NKV_ * HD_) / 128, MROWS_ / 128), 256, 0, stream>>>(
        x, wk, kb, MROWS_, NKV_ * HD_, DIM_);
    gemm_wmma<true, false><<<dim3((NKV_ * HD_) / 128, MROWS_ / 128), 256, 0, stream>>>(
        x, wv, vb, MROWS_, NKV_ * HD_, DIM_);

    // RoPE on q and k, in place
    const long long pairs = (long long)MROWS_ * (NH_ + NKV_) * (HD_ / 2);
    rope_kernel<<<(unsigned)(pairs / 256), 256, 0, stream>>>(qb, kb, freqs);

    // Causal flash attention
    attn_wmma<<<B_ * NH_ * (T_ / 64), 128, 0, stream>>>(qb, kb, vb, yb);

    // Output projection (bf16 in -> f32 out, A staged by TDM)
    gemm_wmma<false, true><<<dim3(DIM_ / 128, MROWS_ / 128), 256, 0, stream>>>(
        yb, wo, out, MROWS_, DIM_, DIM_);
}
